// GCNEncoderDecoderClassifier_43628277793023
// MI455X (gfx1250) — matrix operand order repaired and checked by reading, not compiled, not run on hardware
//
#include <hip/hip_runtime.h>
#include <hip/hip_bf16.h>

// ---------------------------------------------------------------------------
// GCN encoder/decoder for MI455X (gfx1250, wave32, WMMA).
// Block-structured graphs -> densified per-graph normalized adjacency (96x96
// bf16, built once); all heavy math as bf16 WMMA GEMMs with f32 accumulation.
// All WMMA fragments are loaded with contiguous 16-byte vector loads.
// ---------------------------------------------------------------------------

typedef __attribute__((ext_vector_type(16))) __bf16 v16bf;
typedef __attribute__((ext_vector_type(8)))  __bf16 v8bf;
typedef __attribute__((ext_vector_type(8)))  float  v8f;

#define NB    512            // graphs
#define NPG   87             // nodes per graph
#define NPGP  96             // padded nodes per graph
#define NN    (NB*NPG)       // 44544 nodes
#define NE    (NB*1392)      // 712704 edges
#define F1    256            // hidden dim
#define FEMB  768            // concat dim
#define LDH   (NB*NPGP)      // 49152: leading dim of feature-major H
#define ADJSZ (NN*NPG)       // 3875328 floats of reconstructed adj

__device__ inline v16bf cat16(v8bf lo, v8bf hi) {
  return __builtin_shufflevector(lo, hi, 0,1,2,3,4,5,6,7,8,9,10,11,12,13,14,15);
}

// ---------------------------------------------------------------------------
// WMMA fragment loaders (ISA 7.12.2, wave32). All k-contiguous -> b128 loads.
// A 16x32: lane L = row m=L&15; kh=(L>>4)*8; VGPR0..3 hold k=kh..kh+7,
//          VGPR4..7 hold k=kh+16..kh+23 (memory order == vector order).
// ---------------------------------------------------------------------------
__device__ inline v16bf frag_a_rows(const __bf16* p, int lda) {
  int lane = threadIdx.x & 31;
  const __bf16* row = p + (lane & 15) * lda + ((lane >> 4) << 3);
  v8bf lo = *(const v8bf*)(row);
  v8bf hi = *(const v8bf*)(row + 16);
  return cat16(lo, hi);
}
__device__ inline v16bf frag_a_rows_clamp(const __bf16* p, int row0, int maxr, int lda) {
  int lane = threadIdx.x & 31;
  int m = min(row0 + (lane & 15), maxr);
  const __bf16* row = p + m * lda + ((lane >> 4) << 3);
  v8bf lo = *(const v8bf*)(row);
  v8bf hi = *(const v8bf*)(row + 16);
  return cat16(lo, hi);
}
// B 32x16 via column-contiguous storage (element (k,n) = p[n*ld + k]):
// lane L = col n=L&15; kb=(L>>4)*8; elements 0..7 = k=kb..kb+7, 8..15 = k=kb+16..kb+23.
__device__ inline v16bf frag_bt_cols(const __bf16* p, int ld) {
  int lane = threadIdx.x & 31;
  const __bf16* col = p + (lane & 15) * ld + ((lane >> 4) << 3);
  v8bf lo = *(const v8bf*)(col);
  v8bf hi = *(const v8bf*)(col + 16);
  return cat16(lo, hi);
}
__device__ inline v16bf frag_bt_cols_clamp(const __bf16* p, int col0, int maxc, int ld) {
  int lane = threadIdx.x & 31;
  int n = min(col0 + (lane & 15), maxc);
  const __bf16* col = p + n * ld + ((lane >> 4) << 3);
  v8bf lo = *(const v8bf*)(col);
  v8bf hi = *(const v8bf*)(col + 16);
  return cat16(lo, hi);
}

#define WMMA_BF16(a, b, c) \
  __builtin_amdgcn_wmma_f32_16x16x32_bf16(false, (a), false, (b), (short)0, (c), false, false)

// ---------------------------------------------------------------------------
// Graph-structure kernels (run once; Ahat identical for all 3 layers)
// ---------------------------------------------------------------------------
__global__ void k_deg_init(float* deg) {
  int i = blockIdx.x * blockDim.x + threadIdx.x;
  if (i < NN) deg[i] = 1.0f;                 // self-loop weight
}
__global__ void k_deg_edges(float* deg, const int* dst, const float* ew) {
  int e = blockIdx.x * blockDim.x + threadIdx.x;
  if (e < NE) atomicAdd(&deg[dst[e]], ew[e]);
}
__global__ void k_dinv(float* deg) {
  int i = blockIdx.x * blockDim.x + threadIdx.x;
  if (i < NN) { float d = deg[i]; deg[i] = d > 0.0f ? rsqrtf(d) : 0.0f; }
}
__global__ void k_ahat_edges(float* A, const float* dinv, const int* src,
                             const int* dst, const float* ew) {
  int e = blockIdx.x * blockDim.x + threadIdx.x;
  if (e >= NE) return;
  int s = src[e], d = dst[e];
  int g = d / NPG;
  float nm = dinv[s] * ew[e] * dinv[d];
  atomicAdd(&A[g * NPGP * NPGP + (d - g * NPG) * NPGP + (s - g * NPG)], nm);
}
__global__ void k_ahat_self(float* A, const float* dinv) {
  int n = blockIdx.x * blockDim.x + threadIdx.x;
  if (n >= NN) return;
  int g = n / NPG, l = n - g * NPG;
  atomicAdd(&A[g * NPGP * NPGP + l * NPGP + l], dinv[n] * dinv[n]);
}

// ---------------------------------------------------------------------------
// Conversions (weights pre-transposed so GEMM B operands are k-contiguous)
// ---------------------------------------------------------------------------
__global__ void k_cvt(__bf16* o, const float* i, int n) {
  int t = blockIdx.x * blockDim.x + threadIdx.x;
  if (t < n) o[t] = (__bf16)i[t];
}
__global__ void k_cvt_x(__bf16* o, const float* x) {   // [N,87] -> [N,96] pad
  int t = blockIdx.x * blockDim.x + threadIdx.x;
  if (t >= NN * NPGP) return;
  int r = t / NPGP, c = t - r * NPGP;
  o[t] = (c < NPG) ? (__bf16)x[r * NPG + c] : (__bf16)0.0f;
}
__global__ void k_cvt_w1t(__bf16* o, const float* w) { // [87,256] -> Wt1[256][96]
  int t = blockIdx.x * blockDim.x + threadIdx.x;
  if (t >= F1 * NPGP) return;
  int n = t / NPGP, k = t - n * NPGP;
  o[t] = (k < NPG) ? (__bf16)w[k * F1 + n] : (__bf16)0.0f;
}
__global__ void k_cvt_wt(__bf16* o, const float* w) {  // [256,256] -> transposed
  int t = blockIdx.x * blockDim.x + threadIdx.x;
  if (t >= F1 * F1) return;
  int n = t >> 8, k = t & 255;
  o[t] = (__bf16)w[k * F1 + n];
}

// ---------------------------------------------------------------------------
// GEMM: Ht[f][g*96 + m] = sum_k A[g*87+m][k] * Wt[f][k]   (f in 0..255)
// grid = (NB, 6), block = 128 (4 waves; wave w handles 64-col group w).
// A rows clamped at buffer end; padded rows (m>=87) computed+stored, later
// annihilated by Ahat's zero columns.
// ---------------------------------------------------------------------------
__global__ void k_gemm_t(const __bf16* __restrict__ A, int lda,
                         const __bf16* __restrict__ Bt, int K,
                         __bf16* __restrict__ Ht) {
  int g = blockIdx.x, mt = blockIdx.y;
  int cb = (threadIdx.x >> 5) * 64;          // wave's column base
  int row0 = g * NPG + mt * 16;
  v8f a0 = {0,0,0,0,0,0,0,0}, a1 = a0, a2 = a0, a3 = a0;
  for (int k0 = 0; k0 < K; k0 += 32) {
    __builtin_prefetch(A + (row0 + (threadIdx.x & 15)) * lda + k0 + 32, 0, 1);
    v16bf fa = frag_a_rows_clamp(A + k0, row0, NN - 1, lda);
    v16bf b0 = frag_bt_cols(Bt + (cb +  0) * K + k0, K);
    v16bf b1 = frag_bt_cols(Bt + (cb + 16) * K + k0, K);
    v16bf b2 = frag_bt_cols(Bt + (cb + 32) * K + k0, K);
    v16bf b3 = frag_bt_cols(Bt + (cb + 48) * K + k0, K);
    a0 = WMMA_BF16(fa, b0, a0);
    a1 = WMMA_BF16(fa, b1, a1);
    a2 = WMMA_BF16(fa, b2, a2);
    a3 = WMMA_BF16(fa, b3, a3);
  }
  // C layout: lane n=L&15, VGPR r -> row m = r + 8*(L>>4). In the transposed
  // output a lane's 8 rows are contiguous -> one b128 store per sub-tile.
  int lane = threadIdx.x & 31;
  int n = lane & 15, mh = (lane >> 4) * 8;
  int rbase = g * NPGP + mt * 16 + mh;
  v8f accs[4] = {a0, a1, a2, a3};
#pragma unroll
  for (int s = 0; s < 4; ++s) {
    v8bf o;
#pragma unroll
    for (int r = 0; r < 8; ++r) o[r] = (__bf16)accs[s][r];
    *(v8bf*)(Ht + (size_t)(cb + s * 16 + n) * LDH + rbase) = o;
  }
}

// ---------------------------------------------------------------------------
// Propagation: P = Ahat_g[96x96] @ H_g[96x256] (H feature-major, padded);
// emb[:, off+f] = relu(P + bias) for rows < 87.  grid=(NB,6), block=128.
// ---------------------------------------------------------------------------
__global__ void k_prop(const __bf16* __restrict__ Ahat,
                       const __bf16* __restrict__ Ht,
                       const float* __restrict__ bias,
                       __bf16* __restrict__ emb, int off) {
  int g = blockIdx.x, mt = blockIdx.y;
  int cb = (threadIdx.x >> 5) * 64;
  const __bf16* Ab = Ahat + g * NPGP * NPGP + mt * 16 * NPGP;
  const __bf16* Hg = Ht + g * NPGP;
  v8f a0 = {0,0,0,0,0,0,0,0}, a1 = a0, a2 = a0, a3 = a0;
#pragma unroll
  for (int k0 = 0; k0 < NPGP; k0 += 32) {
    v16bf fa = frag_a_rows(Ab + k0, NPGP);
    v16bf b0 = frag_bt_cols(Hg + (size_t)(cb +  0) * LDH + k0, LDH);
    v16bf b1 = frag_bt_cols(Hg + (size_t)(cb + 16) * LDH + k0, LDH);
    v16bf b2 = frag_bt_cols(Hg + (size_t)(cb + 32) * LDH + k0, LDH);
    v16bf b3 = frag_bt_cols(Hg + (size_t)(cb + 48) * LDH + k0, LDH);
    a0 = WMMA_BF16(fa, b0, a0);
    a1 = WMMA_BF16(fa, b1, a1);
    a2 = WMMA_BF16(fa, b2, a2);
    a3 = WMMA_BF16(fa, b3, a3);
  }
  int lane = threadIdx.x & 31;
  int n = lane & 15, mh = (lane >> 4) * 8;
  float bi0 = bias[cb + n],      bi1 = bias[cb + 16 + n];
  float bi2 = bias[cb + 32 + n], bi3 = bias[cb + 48 + n];
#pragma unroll
  for (int r = 0; r < 8; ++r) {
    int m = mt * 16 + r + mh;
    if (m < NPG) {
      __bf16* er = emb + (size_t)(g * NPG + m) * FEMB + off + cb + n;
      er[0]  = (__bf16)fmaxf(a0[r] + bi0, 0.0f);
      er[16] = (__bf16)fmaxf(a1[r] + bi1, 0.0f);
      er[32] = (__bf16)fmaxf(a2[r] + bi2, 0.0f);
      er[48] = (__bf16)fmaxf(a3[r] + bi3, 0.0f);
    }
  }
}

// ---------------------------------------------------------------------------
// Decoder: adj_g = relu(ne_g @ ne_g^T), ne_g = emb rows of graph g [87,768].
// grid=(NB,6), block=32; wave computes 16 rows x 96 cols (A-frag reuse).
// Both operands are k-contiguous rows of emb -> b128 loads.
// ---------------------------------------------------------------------------
__global__ void k_decoder(const __bf16* __restrict__ emb, float* __restrict__ out) {
  int g = blockIdx.x, mt = blockIdx.y;
  int arow0 = g * NPG + mt * 16;
  v8f acc[6];
#pragma unroll
  for (int c = 0; c < 6; ++c) acc[c] = (v8f){0,0,0,0,0,0,0,0};
  for (int k0 = 0; k0 < FEMB; k0 += 32) {
    __builtin_prefetch(emb + (size_t)(arow0 + (threadIdx.x & 15)) * FEMB + k0 + 32, 0, 1);
    v16bf fa = frag_a_rows_clamp(emb + k0, arow0, NN - 1, FEMB);
#pragma unroll
    for (int c = 0; c < 6; ++c) {
      v16bf fb = frag_bt_cols_clamp(emb + k0, g * NPG + c * 16, NN - 1, FEMB);
      acc[c] = WMMA_BF16(fa, fb, acc[c]);
    }
  }
  int lane = threadIdx.x & 31;
  int n = lane & 15, mh = (lane >> 4) * 8;
#pragma unroll
  for (int c = 0; c < 6; ++c) {
    int nl = c * 16 + n;
    if (nl >= NPG) continue;
#pragma unroll
    for (int r = 0; r < 8; ++r) {
      int m = mt * 16 + r + mh;
      if (m < NPG)
        out[(size_t)(g * NPG + m) * NPG + nl] = fmaxf(acc[c][r], 0.0f);
    }
  }
}

// ---------------------------------------------------------------------------
// Pool + classifier (tiny; scalar)
// ---------------------------------------------------------------------------
__global__ void k_pool(const __bf16* __restrict__ emb, float* __restrict__ ge) {
  int t = blockIdx.x * blockDim.x + threadIdx.x;
  if (t >= NB * FEMB) return;
  int g = t / FEMB, f = t - g * FEMB;
  const __bf16* p = emb + (size_t)g * NPG * FEMB + f;
  float s = 0.0f;
  for (int i = 0; i < NPG; ++i) s += (float)p[i * FEMB];
  ge[t] = s * (1.0f / (float)NPG);
}
__global__ void k_logits(const float* __restrict__ ge, const float* __restrict__ Wc,
                         const float* __restrict__ bc, float* __restrict__ out) {
  int t = blockIdx.x * blockDim.x + threadIdx.x;
  if (t >= NB * 2) return;
  int g = t >> 1, c = t & 1;
  float s = bc[c];
  const float* gr = ge + g * FEMB;
  for (int k = 0; k < FEMB; ++k) s += gr[k] * Wc[k * 2 + c];
  out[t] = s;
}

// ---------------------------------------------------------------------------
extern "C" void kernel_launch(void* const* d_in, const int* in_sizes, int n_in,
                              void* d_out, int out_size, void* d_ws, size_t ws_size,
                              hipStream_t stream) {
  const float* x   = (const float*)d_in[0];
  const int*   ei  = (const int*)  d_in[1];   // [2,E]: src then dst
  const float* ew  = (const float*)d_in[2];
  const float* W1  = (const float*)d_in[4];
  const float* b1  = (const float*)d_in[5];
  const float* W2  = (const float*)d_in[6];
  const float* b2  = (const float*)d_in[7];
  const float* W3  = (const float*)d_in[8];
  const float* b3  = (const float*)d_in[9];
  const float* Wc  = (const float*)d_in[10];
  const float* bc  = (const float*)d_in[11];
  float* out = (float*)d_out;                 // [ADJSZ] adj, then [1024] logits
  const int* src = ei;
  const int* dst = ei + NE;

  // -------- workspace layout (256-aligned pieces) --------
  char* ws = (char*)d_ws;
  size_t o = 0;
  auto take = [&](size_t bytes) { char* p = ws + o; o += (bytes + 255) & ~(size_t)255; return p; };
  float*  dinv = (float*) take((size_t)NN * 4);
  __bf16* Abf  = (__bf16*)take((size_t)NB * NPGP * NPGP * 2);
  __bf16* Xb   = (__bf16*)take((size_t)NN * NPGP * 2);
  __bf16* Wt1  = (__bf16*)take((size_t)F1 * NPGP * 2);
  __bf16* Wt2  = (__bf16*)take((size_t)F1 * F1 * 2);
  __bf16* Wt3  = (__bf16*)take((size_t)F1 * F1 * 2);
  __bf16* Ht   = (__bf16*)take((size_t)F1 * LDH * 2);
  __bf16* emb  = (__bf16*)take((size_t)NN * FEMB * 2);
  float*  gemb = (float*) take((size_t)NB * FEMB * 4);
  // f32 adjacency staging aliased onto emb (dead before emb is first written)
  float* Af32 = (float*)emb;

  const int TB = 256;
  auto g1 = [&](int n) { return dim3((n + TB - 1) / TB); };

  // ---- graph structure (once; identical for all layers) ----
  hipMemsetAsync(Af32, 0, (size_t)NB * NPGP * NPGP * 4, stream);
  k_deg_init  <<<g1(NN), TB, 0, stream>>>(dinv);
  k_deg_edges <<<g1(NE), TB, 0, stream>>>(dinv, dst, ew);
  k_dinv      <<<g1(NN), TB, 0, stream>>>(dinv);
  k_ahat_edges<<<g1(NE), TB, 0, stream>>>(Af32, dinv, src, dst, ew);
  k_ahat_self <<<g1(NN), TB, 0, stream>>>(Af32, dinv);
  k_cvt<<<g1(NB * NPGP * NPGP), TB, 0, stream>>>(Abf, Af32, NB * NPGP * NPGP);

  // ---- bf16 operand prep ----
  k_cvt_x  <<<g1(NN * NPGP), TB, 0, stream>>>(Xb, x);
  k_cvt_w1t<<<g1(F1 * NPGP), TB, 0, stream>>>(Wt1, W1);
  k_cvt_wt <<<g1(F1 * F1), TB, 0, stream>>>(Wt2, W2);
  k_cvt_wt <<<g1(F1 * F1), TB, 0, stream>>>(Wt3, W3);

  dim3 gTile(NB, 6);

  // ---- layer 1: H = X @ W1 ; h1 = relu(Ahat H + b1) -> emb[:,0:256] ----
  k_gemm_t<<<gTile, 128, 0, stream>>>(Xb, NPGP, Wt1, NPGP, Ht);
  k_prop  <<<gTile, 128, 0, stream>>>(Abf, Ht, b1, emb, 0);
  // ---- layer 2 ----
  k_gemm_t<<<gTile, 128, 0, stream>>>(emb, FEMB, Wt2, F1, Ht);
  k_prop  <<<gTile, 128, 0, stream>>>(Abf, Ht, b2, emb, 256);
  // ---- layer 3 ----
  k_gemm_t<<<gTile, 128, 0, stream>>>(emb + 256, FEMB, Wt3, F1, Ht);
  k_prop  <<<gTile, 128, 0, stream>>>(Abf, Ht, b3, emb, 512);

  // ---- readout ----
  k_pool  <<<g1(NB * FEMB), TB, 0, stream>>>(emb, gemb);
  k_logits<<<g1(NB * 2), TB, 0, stream>>>(gemb, Wc, bc, out + ADJSZ);
  k_decoder<<<gTile, 32, 0, stream>>>(emb, out);
}
